// AdaptiveDeblurBlock_32942399160874
// MI455X (gfx1250) — matrix operand order repaired and checked by reading, not compiled, not run on hardware
//
#include <hip/hip_runtime.h>
#include <hip/hip_bf16.h>

typedef __attribute__((ext_vector_type(16))) _Float16 v16h;
typedef __attribute__((ext_vector_type(8)))  float    v8f;

// Problem constants
constexpr int Bn = 4, C = 128, H = 192, W = 192;
constexpr int HP = H + 2, WP = W + 2;

enum { OUT_PAD_F16 = 0, OUT_RAW_F16 = 1, OUT_NCHW_F32 = 2 };

// ---------------------------------------------------------------------------
// Pack x: NCHW f32 -> zero-padded NHWC f16. Thread per padded element
// (c fastest => fully coalesced f16 stores; reads are L2-cached NCHW).
// ---------------------------------------------------------------------------
__global__ __launch_bounds__(256) void pack_x_pad(const float* __restrict__ x,
                                                  _Float16* __restrict__ xpad) {
  size_t idx = (size_t)blockIdx.x * blockDim.x + threadIdx.x;
  const size_t total = (size_t)Bn * HP * WP * C;
  if (idx >= total) return;
  int c = (int)(idx % C);
  size_t r = idx / C;
  int wp_ = (int)(r % WP); r /= WP;
  int hp_ = (int)(r % HP);
  int b   = (int)(r / HP);
  float v = 0.f;
  if (hp_ >= 1 && hp_ <= H && wp_ >= 1 && wp_ <= W)
    v = x[((size_t)(b * C + c) * H + (hp_ - 1)) * W + (wp_ - 1)];
  xpad[idx] = (_Float16)v;
}

// Pack weights: OIHW f32 -> [co][tap*CIN+ci] f16 (k-major rows per cout).
template <int CIN, int COUT>
__global__ __launch_bounds__(256) void pack_w(const float* __restrict__ w,
                                              _Float16* __restrict__ wp) {
  int idx = blockIdx.x * blockDim.x + threadIdx.x;
  constexpr int K9 = 9 * CIN;
  if (idx >= COUT * K9) return;
  int co = idx / K9;
  int rem = idx - co * K9;
  int t = rem / CIN;
  int ci = rem - t * CIN;
  wp[idx] = (_Float16)w[(size_t)(co * CIN + ci) * 9 + t];
}

// Zero the 1-px border ring of a padded NHWC f16 buffer.
template <int CC>
__global__ __launch_bounds__(256) void zero_border(_Float16* __restrict__ buf) {
  size_t idx = (size_t)blockIdx.x * blockDim.x + threadIdx.x;
  const size_t total = (size_t)Bn * HP * WP * CC;
  if (idx >= total) return;
  size_t r = idx / CC;
  int wp_ = (int)(r % WP); r /= WP;
  int hp_ = (int)(r % HP);
  if (hp_ == 0 || hp_ == HP - 1 || wp_ == 0 || wp_ == WP - 1)
    buf[idx] = (_Float16)0.f;
}

// ---------------------------------------------------------------------------
// Implicit-GEMM 3x3 conv via v_wmma_f32_16x16x32_f16.
// src: padded NHWC f16 (Bn,HP,WP,CIN). wp: [COUT][9*CIN] f16. bias: f32.
// Each wave owns a 32-pixel x 32-cout macro tile (2x2 WMMA tiles).
// ---------------------------------------------------------------------------
template <int CIN, int COUT, bool RELU, int OUTMODE>
__global__ __launch_bounds__(256) void conv3x3_wmma(
    const _Float16* __restrict__ src, const _Float16* __restrict__ wp,
    const float* __restrict__ bias, void* __restrict__ dstv) {
  constexpr int K9 = 9 * CIN;
  constexpr int NCO = COUT / 32;
  const int lane = threadIdx.x & 31;
  const int wave = (int)((blockIdx.x * blockDim.x + threadIdx.x) >> 5);
  const int co_super = wave % NCO;
  const int p_super = wave / NCO;
  const int p0 = p_super * 32;            // 32 consecutive pixels in one row (32 | W)
  const int b  = p0 / (H * W);
  const int hw = p0 - b * H * W;
  const int h  = hw / W;
  const int w0 = hw - h * W;
  const int m   = lane & 15;              // matrix row / N column within tile
  const int grp = lane >> 4;              // K half-group

  v8f acc[2][2] = {};

  // B-row pointers for the two cout tiles (lane holds 32B of its co row per step)
  const _Float16* wrow0 = wp + (size_t)(co_super * 32 + m) * K9 + grp * 8;
  const _Float16* wrow1 = wrow0 + (size_t)16 * K9;

  for (int t = 0; t < 9; ++t) {
    const int dh = t / 3, dw = t % 3;
    const _Float16* arow0 =
        src + ((size_t)(b * HP + h + dh) * WP + (w0 + m + dw)) * CIN + grp * 8;
    const _Float16* arow1 = arow0 + 16 * CIN;   // second pixel tile (w + 16)
    const _Float16* wk0 = wrow0 + t * CIN;
    const _Float16* wk1 = wrow1 + t * CIN;
#pragma unroll
    for (int cb = 0; cb < CIN; cb += 32) {
      v16h a0, a1, b0, b1;
      ((uint4*)&a0)[0] = *(const uint4*)(arow0 + cb);
      ((uint4*)&a0)[1] = *(const uint4*)(arow0 + cb + 16);
      ((uint4*)&a1)[0] = *(const uint4*)(arow1 + cb);
      ((uint4*)&a1)[1] = *(const uint4*)(arow1 + cb + 16);
      ((uint4*)&b0)[0] = *(const uint4*)(wk0 + cb);
      ((uint4*)&b0)[1] = *(const uint4*)(wk0 + cb + 16);
      ((uint4*)&b1)[0] = *(const uint4*)(wk1 + cb);
      ((uint4*)&b1)[1] = *(const uint4*)(wk1 + cb + 16);
      acc[0][0] = __builtin_amdgcn_wmma_f32_16x16x32_f16(false, a0, false, b0,
                                                         (short)0, acc[0][0], false, false);
      acc[0][1] = __builtin_amdgcn_wmma_f32_16x16x32_f16(false, a0, false, b1,
                                                         (short)0, acc[0][1], false, false);
      acc[1][0] = __builtin_amdgcn_wmma_f32_16x16x32_f16(false, a1, false, b0,
                                                         (short)0, acc[1][0], false, false);
      acc[1][1] = __builtin_amdgcn_wmma_f32_16x16x32_f16(false, a1, false, b1,
                                                         (short)0, acc[1][1], false, false);
    }
  }

  // Epilogue: bias (+ReLU), scatter per C/D layout (VGPR r: M=r+grp*8, N=lane&15)
#pragma unroll
  for (int j = 0; j < 2; ++j) {
    const int co = co_super * 32 + j * 16 + m;
    const float bv = bias[co];
#pragma unroll
    for (int i = 0; i < 2; ++i) {
#pragma unroll
      for (int r = 0; r < 8; ++r) {
        float v = acc[i][j][r] + bv;
        if (RELU) v = v > 0.f ? v : 0.f;
        const int pm = r + grp * 8;
        const int ww = w0 + i * 16 + pm;
        if (OUTMODE == OUT_PAD_F16) {
          ((_Float16*)dstv)[((size_t)(b * HP + h + 1) * WP + (ww + 1)) * COUT + co] =
              (_Float16)v;
        } else if (OUTMODE == OUT_RAW_F16) {
          ((_Float16*)dstv)[((size_t)(b * H + h) * W + ww) * COUT + co] = (_Float16)v;
        } else {
          ((float*)dstv)[((size_t)(b * COUT + co) * H + h) * W + ww] = v;
        }
      }
    }
  }
}

// ---------------------------------------------------------------------------
// conv3 (1x1, 32->9) + softmax + 3x3 per-pixel kernel blend.
// Thread = (pixel, 16-channel group). Logits recomputed per thread (288 MACs).
// ---------------------------------------------------------------------------
__global__ __launch_bounds__(256) void kpn_apply(
    const _Float16* __restrict__ y2,    // (Bn,H,W,32) NHWC f16
    const _Float16* __restrict__ xpad,  // (Bn,HP,WP,128) f16
    const float* __restrict__ w3,       // (9,32)
    const float* __restrict__ b3,       // (9)
    _Float16* __restrict__ outpad) {    // (Bn,HP,WP,128) f16, interior
  const int tid = blockIdx.x * blockDim.x + threadIdx.x;
  if (tid >= Bn * H * W * 8) return;
  const int cg  = tid & 7;
  const int pix = tid >> 3;
  const int b  = pix / (H * W);
  const int hw = pix - b * H * W;
  const int h  = hw / W;
  const int w  = hw - h * W;

  const _Float16* yv = y2 + (size_t)pix * 32;
  float lg[9];
#pragma unroll
  for (int t = 0; t < 9; ++t) {
    float s = b3[t];
#pragma unroll
    for (int ci = 0; ci < 32; ++ci) s += (float)yv[ci] * w3[t * 32 + ci];
    lg[t] = s;
  }
  float mx = lg[0];
#pragma unroll
  for (int t = 1; t < 9; ++t) mx = fmaxf(mx, lg[t]);
  float ssum = 0.f;
#pragma unroll
  for (int t = 0; t < 9; ++t) { lg[t] = __expf(lg[t] - mx); ssum += lg[t]; }
  const float inv = 1.f / ssum;

  const int c0 = cg * 16;
  float accv[16] = {};
#pragma unroll
  for (int t = 0; t < 9; ++t) {
    const int dh = t / 3, dw = t % 3;
    const _Float16* xp =
        xpad + ((size_t)(b * HP + h + dh) * WP + (w + dw)) * C + c0;
    const float kt = lg[t] * inv;
#pragma unroll
    for (int c = 0; c < 16; ++c) accv[c] += kt * (float)xp[c];
  }
  _Float16* op = outpad + ((size_t)(b * HP + h + 1) * WP + (w + 1)) * C + c0;
#pragma unroll
  for (int c = 0; c < 16; ++c) op[c] = (_Float16)accv[c];
}

// ---------------------------------------------------------------------------
extern "C" void kernel_launch(void* const* d_in, const int* in_sizes, int n_in,
                              void* d_out, int out_size, void* d_ws, size_t ws_size,
                              hipStream_t stream) {
  const float* x   = (const float*)d_in[0];
  const float* w1  = (const float*)d_in[1];
  const float* b1  = (const float*)d_in[2];
  const float* w2  = (const float*)d_in[3];
  const float* b2  = (const float*)d_in[4];
  const float* w3  = (const float*)d_in[5];
  const float* b3  = (const float*)d_in[6];
  const float* f1  = (const float*)d_in[7];
  const float* fb1 = (const float*)d_in[8];
  const float* f2  = (const float*)d_in[9];
  const float* fb2 = (const float*)d_in[10];
  float* out = (float*)d_out;

  // Workspace layout (all region sizes are multiples of 256 B)
  char* ws = (char*)d_ws;
  const size_t npadpix = (size_t)Bn * HP * WP;           // 150544
  _Float16* xpad     = (_Float16*)ws;                    ws += npadpix * 128 * 2;
  _Float16* y1pad    = (_Float16*)ws;                    ws += npadpix * 64  * 2;
  _Float16* outpad   = (_Float16*)ws;                    ws += npadpix * 128 * 2;
  _Float16* feat1pad = (_Float16*)ws;                    ws += npadpix * 128 * 2;
  _Float16* y2       = (_Float16*)ws;                    ws += (size_t)Bn * H * W * 32 * 2;
  _Float16* w1p      = (_Float16*)ws;                    ws += (size_t)64  * 1152 * 2;
  _Float16* w2p      = (_Float16*)ws;                    ws += (size_t)32  * 576  * 2;
  _Float16* f1p      = (_Float16*)ws;                    ws += (size_t)128 * 1152 * 2;
  _Float16* f2p      = (_Float16*)ws;                    ws += (size_t)128 * 1152 * 2;
  (void)ws_size; (void)in_sizes; (void)n_in; (void)out_size;

  // 1) Packing / padding
  {
    const size_t n = npadpix * 128;
    pack_x_pad<<<(int)((n + 255) / 256), 256, 0, stream>>>(x, xpad);
  }
  pack_w<128, 64><<<(64 * 1152 + 255) / 256, 256, 0, stream>>>(w1, w1p);
  pack_w<64, 32><<<(32 * 576 + 255) / 256, 256, 0, stream>>>(w2, w2p);
  pack_w<128, 128><<<(128 * 1152 + 255) / 256, 256, 0, stream>>>(f1, f1p);
  pack_w<128, 128><<<(128 * 1152 + 255) / 256, 256, 0, stream>>>(f2, f2p);
  zero_border<64><<<(int)((npadpix * 64 + 255) / 256), 256, 0, stream>>>(y1pad);
  zero_border<128><<<(int)((npadpix * 128 + 255) / 256), 256, 0, stream>>>(outpad);
  zero_border<128><<<(int)((npadpix * 128 + 255) / 256), 256, 0, stream>>>(feat1pad);

  const int pixel_supertiles = Bn * H * W / 32;  // 4608

  // 2) conv1: 128 -> 64, ReLU, padded f16 out
  {
    int waves = pixel_supertiles * (64 / 32);
    conv3x3_wmma<128, 64, true, OUT_PAD_F16>
        <<<waves / 8, 256, 0, stream>>>(xpad, w1p, b1, (void*)y1pad);
  }
  // 3) conv2: 64 -> 32, ReLU, raw NHWC f16 out
  {
    int waves = pixel_supertiles * (32 / 32);
    conv3x3_wmma<64, 32, true, OUT_RAW_F16>
        <<<waves / 8, 256, 0, stream>>>(y1pad, w2p, b2, (void*)y2);
  }
  // 4) conv3(1x1) + softmax + per-pixel 3x3 blend
  {
    int n = Bn * H * W * 8;
    kpn_apply<<<(n + 255) / 256, 256, 0, stream>>>(y2, xpad, w3, b3, outpad);
  }
  // 5) f1: 128 -> 128, ReLU, padded f16 out
  {
    int waves = pixel_supertiles * (128 / 32);
    conv3x3_wmma<128, 128, true, OUT_PAD_F16>
        <<<waves / 8, 256, 0, stream>>>(outpad, f1p, fb1, (void*)feat1pad);
  }
  // 6) f2: 128 -> 128, no ReLU, NCHW f32 -> d_out
  {
    int waves = pixel_supertiles * (128 / 32);
    conv3x3_wmma<128, 128, false, OUT_NCHW_F32>
        <<<waves / 8, 256, 0, stream>>>(feat1pad, f2p, fb2, (void*)out);
  }
}
